// TransformerDecoder_7954279432673
// MI455X (gfx1250) — matrix operand order repaired and checked by reading, not compiled, not run on hardware
//
#include <hip/hip_runtime.h>

// Problem constants (match reference)
#define LL 6
#define DD 256
#define HH 8
#define DH 32
#define FF 1024
#define VV 3000
#define SS 512
#define MM 1024
#define BB 16

typedef __attribute__((ext_vector_type(16))) _Float16 v16h;
typedef __attribute__((ext_vector_type(8)))  float    v8f;

union FragH {
    v16h   v;
    float4 f4[2];
};

// ---------------------------------------------------------------------------
// f32 -> f16 convert (grid-stride)
// ---------------------------------------------------------------------------
__global__ void convert_f16_kernel(const float* __restrict__ in,
                                   _Float16* __restrict__ out, size_t n) {
    size_t i = (size_t)blockIdx.x * blockDim.x + threadIdx.x;
    size_t stride = (size_t)gridDim.x * blockDim.x;
    for (; i < n; i += stride) out[i] = (_Float16)in[i];
}

// ---------------------------------------------------------------------------
// Batched f32 [K,N] -> f16 [N,K] transpose-convert (one-time per launch).
// Output index i = (b, n, k); writes coalesced.
// ---------------------------------------------------------------------------
__global__ void transpose_cvt_kernel(const float* __restrict__ in,
                                     _Float16* __restrict__ out,
                                     int batch, int K, int N) {
    size_t total = (size_t)batch * K * N;
    size_t i = (size_t)blockIdx.x * blockDim.x + threadIdx.x;
    size_t stride = (size_t)gridDim.x * blockDim.x;
    for (; i < total; i += stride) {
        size_t mat = i / ((size_t)K * N);
        size_t r   = i % ((size_t)K * N);
        int n = (int)(r / K);
        int k = (int)(r % K);
        out[i] = (_Float16)in[mat * K * N + (size_t)k * N + n];
    }
}

// ---------------------------------------------------------------------------
// Embedding: x[b,s,:] = input_embed[seq[b,s],:] + pos_embed[s,:]
// ---------------------------------------------------------------------------
__global__ __launch_bounds__(256)
void embed_kernel(const int* __restrict__ seq, const float* __restrict__ emb,
                  const float* __restrict__ pos, float* __restrict__ x32,
                  _Float16* __restrict__ xh) {
    int row = blockIdx.x;          // b*S + s
    int s   = row % SS;
    int d   = threadIdx.x;
    int tok = seq[row];
    float v = emb[(size_t)tok * DD + d] + pos[(size_t)s * DD + d];
    size_t idx = (size_t)row * DD + d;
    x32[idx] = v;
    xh[idx]  = (_Float16)v;
}

// ---------------------------------------------------------------------------
// WMMA f16 GEMM (NT): C[M x N] = A[M x K] * Bt^T   (+bias, opt relu)
//   A : f16 row-major [M,K]       (M % 128 == 0 assumed for loads)
//   Bt: f16 row-major [N,K]       (rows beyond N must be readable: slack)
//   Cf (f32) and/or Ch (f16) outputs, either may be null
//
// Block: 256 threads (8 waves). Block tile 128x64, wave tile 32x32
// (4 accumulators, 4 WMMAs per k-step). K-step 32, double-buffered LDS
// filled with GLOBAL_LOAD_ASYNC_TO_LDS_B128 + s_wait_asynccnt pipeline.
// LDS rows padded to 40 halves (80B, 16B aligned) for conflict-free b128.
// ---------------------------------------------------------------------------
__global__ __launch_bounds__(256)
void gemm_f16_kernel(const _Float16* __restrict__ A,
                     const _Float16* __restrict__ Bt,
                     float* __restrict__ Cf, _Float16* __restrict__ Ch,
                     const float* __restrict__ bias,
                     int Mtot, int N, int K, int relu) {
    __shared__ _Float16 Al[2][128 * 40];
    __shared__ _Float16 Bl[2][64 * 40];

    const int tid    = threadIdx.x;
    const int lane   = tid & 31;
    const int wid    = tid >> 5;
    const int waveM  = (wid & 3) * 32;
    const int waveN  = (wid >> 2) * 32;
    const int mBase  = blockIdx.y * 128;
    const int nBase  = blockIdx.x * 64;
    const int lm     = lane & 15;
    const int hiHalf = lane >> 4;

    // staging coordinates: A = 128 rows x 32 k (2 x b128/thread),
    //                      B =  64 rows x 32 k (1 x b128/thread)
    const int ar0 = tid >> 2;              // 0..63
    const int ar1 = ar0 + 64;              // 64..127
    const int ac  = (tid & 3) * 8;         // k offset (halves), 16B aligned
    const int br  = tid >> 2;              // 0..63

    const _Float16* gA0 = A  + (size_t)(mBase + ar0) * K + ac;
    const _Float16* gA1 = A  + (size_t)(mBase + ar1) * K + ac;
    const _Float16* gB  = Bt + (size_t)(nBase + br)  * K + ac;

    v8f acc00 = {}, acc01 = {}, acc10 = {}, acc11 = {};

    const int nTiles = K >> 5;

    auto issue = [&](int t, int buf) {
        int kb = t << 5;
        unsigned la0 = (unsigned)(size_t)&Al[buf][ar0 * 40 + ac];
        unsigned la1 = (unsigned)(size_t)&Al[buf][ar1 * 40 + ac];
        unsigned lb  = (unsigned)(size_t)&Bl[buf][br  * 40 + ac];
        unsigned long long ga0 = (unsigned long long)(size_t)(gA0 + kb);
        unsigned long long ga1 = (unsigned long long)(size_t)(gA1 + kb);
        unsigned long long gb  = (unsigned long long)(size_t)(gB  + kb);
        asm volatile(
            "global_load_async_to_lds_b128 %0, %3, off\n\t"
            "global_load_async_to_lds_b128 %1, %4, off\n\t"
            "global_load_async_to_lds_b128 %2, %5, off\n\t"
            "s_wait_xcnt 0x0"
            :: "v"(la0), "v"(la1), "v"(lb), "v"(ga0), "v"(ga1), "v"(gb)
            : "memory");
    };

    issue(0, 0);

    for (int t = 0; t < nTiles; ++t) {
        const int cur = t & 1;
        if (t + 1 < nTiles) {
            issue(t + 1, cur ^ 1);
            // 3 loads of the next tile outstanding -> current tile complete
            asm volatile("s_wait_asynccnt 0x3" ::: "memory");
        } else {
            asm volatile("s_wait_asynccnt 0x0" ::: "memory");
        }
        __syncthreads();

        const _Float16* Ab = Al[cur];
        const _Float16* Bb = Bl[cur];
        FragH a0, a1, b0, b1;
        {
            int klo = hiHalf ? 8 : 0;   // A lanes 16-31 hold K=8..15 / 24..31
            int r0 = waveM + lm;
            a0.f4[0] = *(const float4*)(&Ab[r0 * 40 + klo]);
            a0.f4[1] = *(const float4*)(&Ab[r0 * 40 + klo + 16]);
            int r1 = r0 + 16;
            a1.f4[0] = *(const float4*)(&Ab[r1 * 40 + klo]);
            a1.f4[1] = *(const float4*)(&Ab[r1 * 40 + klo + 16]);
        }
        {
            int kk = hiHalf ? 8 : 0;    // B lanes 16-31 hold K=8..15 / 24..31
            int n0 = waveN + lm;
            b0.f4[0] = *(const float4*)(&Bb[n0 * 40 + kk]);
            b0.f4[1] = *(const float4*)(&Bb[n0 * 40 + kk + 16]);
            int n1 = n0 + 16;
            b1.f4[0] = *(const float4*)(&Bb[n1 * 40 + kk]);
            b1.f4[1] = *(const float4*)(&Bb[n1 * 40 + kk + 16]);
        }
        acc00 = __builtin_amdgcn_wmma_f32_16x16x32_f16(
            false, a0.v, false, b0.v, (short)0, acc00, false, false);
        acc01 = __builtin_amdgcn_wmma_f32_16x16x32_f16(
            false, a0.v, false, b1.v, (short)0, acc01, false, false);
        acc10 = __builtin_amdgcn_wmma_f32_16x16x32_f16(
            false, a1.v, false, b0.v, (short)0, acc10, false, false);
        acc11 = __builtin_amdgcn_wmma_f32_16x16x32_f16(
            false, a1.v, false, b1.v, (short)0, acc11, false, false);
        __syncthreads();
    }

    // epilogue: C layout: VGPR j -> M = j (+8 for lanes 16-31), col = lane%16
#pragma unroll
    for (int mt = 0; mt < 2; ++mt) {
#pragma unroll
        for (int nt = 0; nt < 2; ++nt) {
            v8f acc = mt ? (nt ? acc11 : acc10) : (nt ? acc01 : acc00);
            int col = nBase + waveN + nt * 16 + lm;
            if (col >= N) continue;
            float bv = bias ? bias[col] : 0.f;
#pragma unroll
            for (int j = 0; j < 8; ++j) {
                int row = mBase + waveM + mt * 16 + hiHalf * 8 + j;
                if (row >= Mtot) continue;
                float val = acc[j] + bv;
                if (relu) val = fmaxf(val, 0.f);
                size_t idx = (size_t)row * N + col;
                if (Cf) Cf[idx] = val;
                if (Ch) Ch[idx] = (_Float16)val;
            }
        }
    }
}

// ---------------------------------------------------------------------------
// Flash-style attention: one wave per query, lane = head-dim element (DH=32).
// Streams keys with online softmax; never materializes the score matrix.
// ---------------------------------------------------------------------------
__global__ __launch_bounds__(256)
void attn_kernel(const float* __restrict__ Q, const float* __restrict__ Km,
                 const float* __restrict__ Vm, _Float16* __restrict__ O,
                 int Sq, int nK, int causal) {
    int lane = threadIdx.x & 31;
    int wid  = threadIdx.x >> 5;
    int qi   = blockIdx.x * 8 + wid;
    int h    = blockIdx.y;
    int b    = blockIdx.z;

    size_t qoff = ((size_t)b * Sq + qi) * DD + h * DH + lane;
    float q = Q[qoff] * 0.17677669529663687f;     // 1/sqrt(32)
    size_t kbase = (size_t)b * nK * DD + h * DH + lane;
    int lim = causal ? (qi + 1) : nK;

    float m = -1e30f, l = 0.f, acc = 0.f;
    for (int key = 0; key < lim; ++key) {
        float s = q * Km[kbase + (size_t)key * DD];
        s += __shfl_xor(s, 16, 32);
        s += __shfl_xor(s, 8, 32);
        s += __shfl_xor(s, 4, 32);
        s += __shfl_xor(s, 2, 32);
        s += __shfl_xor(s, 1, 32);
        float mn = fmaxf(m, s);
        float p  = __expf(s - mn);
        float c  = __expf(m - mn);
        l   = l * c + p;
        acc = acc * c + p * Vm[kbase + (size_t)key * DD];
        m = mn;
    }
    O[qoff] = (_Float16)(acc / l);
}

// ---------------------------------------------------------------------------
// Fused residual-add + LayerNorm; emits f32 (residual stream, in-place safe)
// and f16 (next GEMM operand). One block per row, 256 threads = D.
// ---------------------------------------------------------------------------
__global__ __launch_bounds__(256)
void ln_kernel(const float* __restrict__ X, const float* __restrict__ R,
               const float* __restrict__ g, const float* __restrict__ bta,
               float* __restrict__ Xo, _Float16* __restrict__ Xh) {
    __shared__ float buf[256];
    int row = blockIdx.x;
    int d   = threadIdx.x;
    size_t idx = (size_t)row * DD + d;

    float r = X[idx] + (R ? R[idx] : 0.f);
    buf[d] = r;
    __syncthreads();
    for (int s = 128; s > 0; s >>= 1) {
        if (d < s) buf[d] += buf[d + s];
        __syncthreads();
    }
    float mu = buf[0] * (1.f / DD);
    __syncthreads();
    float dv = r - mu;
    buf[d] = dv * dv;
    __syncthreads();
    for (int s = 128; s > 0; s >>= 1) {
        if (d < s) buf[d] += buf[d + s];
        __syncthreads();
    }
    float var = buf[0] * (1.f / DD);
    float y = dv * rsqrtf(var + 1e-5f) * g[d] + bta[d];
    Xo[idx] = y;
    Xh[idx] = (_Float16)y;
}

// ---------------------------------------------------------------------------
// Orchestration
// ---------------------------------------------------------------------------
extern "C" void kernel_launch(void* const* d_in, const int* in_sizes, int n_in,
                              void* d_out, int out_size, void* d_ws, size_t ws_size,
                              hipStream_t stream) {
    (void)in_sizes; (void)n_in; (void)out_size; (void)ws_size;

    const float* encoded = (const float*)d_in[0];
    const int*   seq     = (const int*)d_in[1];
    const float* emb     = (const float*)d_in[2];
    const float* pos     = (const float*)d_in[3];
    const float* obias   = (const float*)d_in[4];
    const float* sa_qkv  = (const float*)d_in[5];
    const float* sa_o    = (const float*)d_in[6];
    const float* ca_q    = (const float*)d_in[7];
    const float* ca_kv   = (const float*)d_in[8];
    const float* ca_o    = (const float*)d_in[9];
    const float* ffn_w1  = (const float*)d_in[10];
    const float* ffn_b1  = (const float*)d_in[11];
    const float* ffn_w2  = (const float*)d_in[12];
    const float* ffn_b2  = (const float*)d_in[13];
    const float* ln_g    = (const float*)d_in[14];
    const float* ln_b    = (const float*)d_in[15];
    const float* fin_g   = (const float*)d_in[16];
    const float* fin_b   = (const float*)d_in[17];

    // ---- workspace carving (256B aligned) ----
    char* ws = (char*)d_ws;
    size_t off = 0;
    auto alloc = [&](size_t bytes) -> void* {
        void* p = ws + off;
        off += (bytes + 255) & ~(size_t)255;
        return p;
    };
    const size_t nBSD = (size_t)BB * SS * DD;   // 2,097,152
    const size_t nBMD = (size_t)BB * MM * DD;   // 4,194,304

    float*    x32  = (float*)   alloc(nBSD * 4);           // residual stream
    _Float16* xh   = (_Float16*)alloc(nBSD * 2);           // f16 activations
    _Float16* ench = (_Float16*)alloc(nBMD * 2);           // f16 encoder memory
    float*    q32  = (float*)   alloc(nBSD * 4);
    float*    k32  = (float*)   alloc(nBMD * 4);
    float*    v32  = (float*)   alloc(nBMD * 4);
    _Float16* aoh  = (_Float16*)alloc(nBSD * 2);           // attn out (f16)
    float*    p32  = (float*)   alloc(nBSD * 4);           // projection out
    _Float16* hh   = (_Float16*)alloc((size_t)BB * SS * FF * 2); // ffn hidden
    // weight pool (+32768 elem slack: ragged-N tiles may read past w_emb end)
    _Float16* wpool = (_Float16*)alloc(((size_t)7059456 + 32768) * 2);

    // weight pool sub-offsets (element units); all stored [N,K] f16
    _Float16* w_sa_qkv = wpool;                    // 18 x [D,D]^T
    _Float16* w_sa_o   = wpool + 1179648;          //  6 x [D,D]^T
    _Float16* w_ca_q   = wpool + 1572864;
    _Float16* w_ca_kv  = wpool + 1966080;          // 12 x [D,D]^T
    _Float16* w_ca_o   = wpool + 2752512;
    _Float16* w_f1     = wpool + 3145728;          //  6 x [F,D]   ([D,F]^T)
    _Float16* w_f2     = wpool + 4718592;          //  6 x [D,F]   ([F,D]^T)
    _Float16* w_emb    = wpool + 6291456;          //  [V,D] (already NT)

    auto cvt = [&](const float* src, _Float16* dst, size_t n) {
        int blocks = (int)((n + 2047) / 2048);
        if (blocks > 4096) blocks = 4096;
        convert_f16_kernel<<<blocks, 256, 0, stream>>>(src, dst, n);
    };
    auto tcvt = [&](const float* src, _Float16* dst, int batch, int K, int N) {
        size_t n = (size_t)batch * K * N;
        int blocks = (int)((n + 2047) / 2048);
        if (blocks > 4096) blocks = 4096;
        transpose_cvt_kernel<<<blocks, 256, 0, stream>>>(src, dst, batch, K, N);
    };
    tcvt(sa_qkv,  w_sa_qkv, 18, DD, DD);
    tcvt(sa_o,    w_sa_o,    6, DD, DD);
    tcvt(ca_q,    w_ca_q,    6, DD, DD);
    tcvt(ca_kv,   w_ca_kv,  12, DD, DD);
    tcvt(ca_o,    w_ca_o,    6, DD, DD);
    tcvt(ffn_w1,  w_f1,      6, DD, FF);
    tcvt(ffn_w2,  w_f2,      6, FF, DD);
    cvt(emb,      w_emb,  768000);
    cvt(encoded,  ench,   nBMD);

    const int TQ = BB * SS;   // 8192 token rows
    embed_kernel<<<TQ, 256, 0, stream>>>(seq, emb, pos, x32, xh);

    auto gemm = [&](const _Float16* A, const _Float16* Bt, float* Cf,
                    _Float16* Ch, const float* bias, int Mt, int N, int K,
                    int relu) {
        dim3 g((N + 63) / 64, (Mt + 127) / 128);
        gemm_f16_kernel<<<g, 256, 0, stream>>>(A, Bt, Cf, Ch, bias, Mt, N, K,
                                               relu);
    };

    for (int i = 0; i < LL; ++i) {
        // ---- self-attention (causal) ----
        const _Float16* Wq = w_sa_qkv + (size_t)i * 3 * DD * DD;
        gemm(xh, Wq,               q32, nullptr, nullptr, TQ, DD, DD, 0);
        gemm(xh, Wq + DD * DD,     k32, nullptr, nullptr, TQ, DD, DD, 0);
        gemm(xh, Wq + 2 * DD * DD, v32, nullptr, nullptr, TQ, DD, DD, 0);
        attn_kernel<<<dim3(SS / 8, HH, BB), 256, 0, stream>>>(
            q32, k32, v32, aoh, SS, SS, 1);
        gemm(aoh, w_sa_o + (size_t)i * DD * DD, p32, nullptr, nullptr,
             TQ, DD, DD, 0);
        ln_kernel<<<TQ, 256, 0, stream>>>(x32, p32,
                                          ln_g + ((size_t)i * 3 + 0) * DD,
                                          ln_b + ((size_t)i * 3 + 0) * DD,
                                          x32, xh);

        // ---- cross-attention over encoder memory ----
        const _Float16* Wkv = w_ca_kv + (size_t)i * 2 * DD * DD;
        gemm(xh,   w_ca_q + (size_t)i * DD * DD, q32, nullptr, nullptr,
             TQ, DD, DD, 0);
        gemm(ench, Wkv,           k32, nullptr, nullptr, BB * MM, DD, DD, 0);
        gemm(ench, Wkv + DD * DD, v32, nullptr, nullptr, BB * MM, DD, DD, 0);
        attn_kernel<<<dim3(SS / 8, HH, BB), 256, 0, stream>>>(
            q32, k32, v32, aoh, SS, MM, 0);
        gemm(aoh, w_ca_o + (size_t)i * DD * DD, p32, nullptr, nullptr,
             TQ, DD, DD, 0);
        ln_kernel<<<TQ, 256, 0, stream>>>(x32, p32,
                                          ln_g + ((size_t)i * 3 + 1) * DD,
                                          ln_b + ((size_t)i * 3 + 1) * DD,
                                          x32, xh);

        // ---- FFN: relu(x@W1+b1)@W2+b2 ----
        gemm(xh, w_f1 + (size_t)i * DD * FF, nullptr, hh,
             ffn_b1 + (size_t)i * FF, TQ, FF, DD, 1);
        gemm(hh, w_f2 + (size_t)i * FF * DD, p32, nullptr,
             ffn_b2 + (size_t)i * DD, TQ, DD, FF, 0);
        ln_kernel<<<TQ, 256, 0, stream>>>(x32, p32,
                                          ln_g + ((size_t)i * 3 + 2) * DD,
                                          ln_b + ((size_t)i * 3 + 2) * DD,
                                          x32, xh);
    }

    // final norm + tied-embedding output projection (NT GEMM, N = 3000)
    ln_kernel<<<TQ, 256, 0, stream>>>(x32, nullptr, fin_g, fin_b, x32, xh);
    gemm(xh, w_emb, (float*)d_out, nullptr, obias, TQ, VV, DD, 0);
}
